// WKVAttention_40845138985319
// MI455X (gfx1250) — compile-verified, matrix operand order checked by reading, not run on hardware
//
#include <hip/hip_runtime.h>

// ---------------- types ----------------
typedef __attribute__((ext_vector_type(16))) __bf16 v16bf;
typedef __attribute__((ext_vector_type(2)))  __bf16 v2bf;
typedef __attribute__((ext_vector_type(8)))  float  v8f;

#define DD 128
#define TT 4096
#define BB 8
#define CC 64
#define LL 64
#define MT (BB*TT)    // 32768 rows
#define NBC (BB*CC)   // 512 (batch,chunk) pairs

__device__ __forceinline__ __bf16 f2bf(float f) { return (__bf16)f; }
__device__ __forceinline__ float frcp(float f) { return __builtin_amdgcn_rcpf(f); }

__device__ __forceinline__ v8f wmma_bf16(v16bf a, v16bf b, v8f c) {
  return __builtin_amdgcn_wmma_f32_16x16x32_bf16(false, a, false, b, (short)0, c, false, false);
}

// ---- fragment-swizzle helpers (ISA 7.12.2 16-bit operand layout, wave32) ----
// within a K-block of 32: k_local -> (hi: lane-half bit, idx: element 0..15)
__device__ __forceinline__ void kdec(int k_local, int& hi, int& idx) {
  hi = (k_local >> 3) & 1;
  int v = ((k_local >> 4) << 2) | ((k_local & 7) >> 1);
  idx = 2 * v + (k_local & 1);
}
// load one 32B fragment (16 bf16) that was stored contiguously per lane
__device__ __forceinline__ v16bf frag_ld(const __bf16* p) {
  return *(const v16bf*)p;          // -> 2x b128 loads
}
// A fragment gathered directly from a row-major f32 row (converted to bf16)
__device__ __forceinline__ v16bf fragA_f32_row(const float* row, int k0, int lane) {
  const int h = (lane & 16) ? 8 : 0;
  v16bf a;
#pragma unroll
  for (int j = 0; j < 8; ++j) {
    a[j]     = f2bf(row[k0 + h + j]);        // 8 consecutive f32 = 2x b128
    a[8 + j] = f2bf(row[k0 + 16 + h + j]);
  }
  return a;
}
// A fragment gathered directly from a row-major bf16 row
__device__ __forceinline__ v16bf fragA_bf16_row(const __bf16* row, int k0, int lane) {
  const int h = (lane & 16) ? 8 : 0;
  v16bf a;
#pragma unroll
  for (int j = 0; j < 8; ++j) {
    a[j]     = row[k0 + h + j];              // 8 consecutive bf16 = b128
    a[8 + j] = row[k0 + 16 + h + j];
  }
  return a;
}

// =====================================================================
// K0: swizzle the 5 weight matrices into bf16 B-fragment layout.
// B[k][n] = W[n][k]; tile set: [kt(4)][nt(8)][lane(32)][16]
// =====================================================================
__global__ __launch_bounds__(256) void k_swizw(
    const float* __restrict__ Wq, const float* __restrict__ Wk,
    const float* __restrict__ Wv, const float* __restrict__ Wd,
    const float* __restrict__ Wo, __bf16* __restrict__ wsw) {
  const int which = blockIdx.x;
  const float* W = (which == 0) ? Wq : (which == 1) ? Wk : (which == 2) ? Wv
                 : (which == 3) ? Wd : Wo;
  __bf16* dst = wsw + (size_t)which * 16384;
  for (int pe = threadIdx.x; pe < 128 * 64; pe += 256) {
    int n = pe >> 6, k = (pe & 63) * 2;      // k,k+1 contiguous in W row
    float2 w2 = *(const float2*)&W[n * DD + k];
    int hi, idx; kdec(k & 31, hi, idx);
    int off = (((k >> 5) * 8 + (n >> 4)) * 32 + ((hi << 4) | (n & 15))) * 16 + idx;
    *(v2bf*)&dst[off] = v2bf{f2bf(w2.x), f2bf(w2.y)};
  }
}

// =====================================================================
// K1: fused projections out = x @ W^T (blockIdx.y: q,k,v,w+sigmoid)
// 8 waves, 128x128 tile; A straight from global f32, B from swizzled weights.
// =====================================================================
__global__ __launch_bounds__(256) void k_proj(
    const float* __restrict__ x, const __bf16* __restrict__ wsw,
    float* __restrict__ qf, float* __restrict__ kf,
    float* __restrict__ vf, float* __restrict__ wf) {
  const int which = blockIdx.y;
  float* out = (which == 0) ? qf : (which == 1) ? kf : (which == 2) ? vf : wf;
  const __bf16* wb = wsw + (size_t)which * 16384;
  const int lane = threadIdx.x & 31, wv = threadIdx.x >> 5;
  const size_t m0 = (size_t)blockIdx.x * 128;
  const size_t mrow = m0 + wv * 16 + (lane & 15);
  const float* row = x + mrow * DD;
  __builtin_prefetch(row + (size_t)128 * DD, 0, 1);      // next block's rows

  v16bf a4[4];
#pragma unroll
  for (int kt = 0; kt < 4; ++kt) a4[kt] = fragA_f32_row(row, kt * 32, lane);
#pragma unroll
  for (int nt = 0; nt < 8; ++nt) {
    v8f acc = {};
#pragma unroll
    for (int kt = 0; kt < 4; ++kt)
      acc = wmma_bf16(a4[kt], frag_ld(wb + ((kt * 8 + nt) * 32 + lane) * 16), acc);
#pragma unroll
    for (int r = 0; r < 8; ++r) {
      int rrow = wv * 16 + r + ((lane & 16) ? 8 : 0);
      int col = nt * 16 + (lane & 15);
      float v = acc[r];
      if (which == 3) v = frcp(1.0f + __expf(-v));   // sigmoid via v_rcp_f32
      out[(m0 + rrow) * DD + col] = v;
    }
  }
}

// =====================================================================
// K2: cumulative decay; emit all chunk operands pre-swizzled (bf16):
//  qb/qtb: A-layout (M=t 4 tiles, K=i 4 tiles)
//  ktb:    B-layout (K=i 4 tiles, N=t 4 tiles)
//  khb:    A-layout (M=i 8 tiles, K=t 2 tiles)
//  vb:     B-layout (K=t 2 tiles, N=i 8 tiles)
// each chunk block = 8192 bf16
// =====================================================================
__global__ __launch_bounds__(128) void k_decay(
    const float* __restrict__ qf, const float* __restrict__ kf,
    const float* __restrict__ vf, const float* __restrict__ wf,
    __bf16* __restrict__ qb, __bf16* __restrict__ qtb, __bf16* __restrict__ ktb,
    __bf16* __restrict__ khb, __bf16* __restrict__ vb, float* __restrict__ pl) {
  const int blk = blockIdx.x;                 // b*CC + c
  const int i = threadIdx.x;                  // dim
  const size_t base = (size_t)blk * LL * DD + i;
  const size_t cb = (size_t)blk * 8192;
  int hi_i, idx_i; kdec(i & 31, hi_i, idx_i);            // K = i decomposition
  const int kt_i = i >> 5, mt_i = i >> 4, i15 = i & 15;
  float P = 1.f;
#pragma unroll 4
  for (int t = 0; t < LL; ++t) {
    size_t idx = base + (size_t)t * DD;
    P *= wf[idx];
    float q = qf[idx], k = kf[idx];
    int hi_t, idx_t; kdec(t & 31, hi_t, idx_t);          // K = t decomposition
    const int mt_t = t >> 4, kt_t = t >> 5, t15 = t & 15;
    int offA = ((mt_t * 4 + kt_i) * 32 + ((hi_i << 4) | t15)) * 16 + idx_i;
    qb[cb + offA]  = f2bf(q);
    qtb[cb + offA] = f2bf(q * P);
    ktb[cb + ((kt_i * 4 + mt_t) * 32 + ((hi_i << 4) | t15)) * 16 + idx_i] =
        f2bf(k * frcp(P));                               // k/P via v_rcp_f32
    vb[cb + ((kt_t * 8 + mt_i) * 32 + ((hi_t << 4) | i15)) * 16 + idx_t]  = f2bf(vf[idx]);
  }
  const float PLv = P;
  pl[(size_t)blk * DD + i] = PLv;
  float P2 = 1.f;
#pragma unroll 4
  for (int t = 0; t < LL; ++t) {
    size_t idx = base + (size_t)t * DD;
    P2 *= wf[idx];
    int hi_t, idx_t; kdec(t & 31, hi_t, idx_t);
    khb[cb + ((mt_i * 2 + (t >> 5)) * 32 + ((hi_t << 4) | i15)) * 16 + idx_t] =
        f2bf(kf[idx] * (PLv * frcp(P2)));                // PL/P2 via v_rcp_f32
  }
}

// =====================================================================
// K3: chunk-final states  Sf[b,c] = K̂ᵀ V  (128x128, K=64); all operands
// arrive fragment-swizzled from global — no LDS, no barriers.
// =====================================================================
__global__ __launch_bounds__(256) void k_chunk_state(
    const __bf16* __restrict__ khb, const __bf16* __restrict__ vb,
    float* __restrict__ sf) {
  const int lane = threadIdx.x & 31, wv = threadIdx.x >> 5;  // wv = m-tile
  const size_t cb = (size_t)blockIdx.x * 8192;
  v16bf a2[2];
#pragma unroll
  for (int kt = 0; kt < 2; ++kt)
    a2[kt] = frag_ld(khb + cb + ((wv * 2 + kt) * 32 + lane) * 16);
  float* out = sf + (size_t)blockIdx.x * DD * DD;
#pragma unroll
  for (int nt = 0; nt < 8; ++nt) {
    v8f acc = {};
#pragma unroll
    for (int kt = 0; kt < 2; ++kt)
      acc = wmma_bf16(a2[kt], frag_ld(vb + cb + ((kt * 8 + nt) * 32 + lane) * 16), acc);
#pragma unroll
    for (int r = 0; r < 8; ++r) {
      int row = wv * 16 + r + ((lane & 16) ? 8 : 0);
      out[row * DD + nt * 16 + (lane & 15)] = acc[r];
    }
  }
}

// =====================================================================
// K4: inter-chunk prefix scan; writes S_prefix directly in bf16 B-layout
// (K=i 4 tiles, N=j 8 tiles; chunk block = 16384 bf16)
// =====================================================================
__global__ __launch_bounds__(256) void k_inter(
    const float* __restrict__ sf, const float* __restrict__ pl,
    __bf16* __restrict__ spb) {
  const int tg = blockIdx.x * 256 + threadIdx.x;   // B*D*D
  const int b = tg >> 14, ij = tg & 16383;
  const int i = ij >> 7, j = ij & 127;
  int hi, idx; kdec(i & 31, hi, idx);
  const int soff = (((i >> 5) * 8 + (j >> 4)) * 32 + ((hi << 4) | (j & 15))) * 16 + idx;
  float S = 0.f;
  for (int c = 0; c < CC; ++c) {
    spb[((size_t)(b * CC + c) << 14) + soff] = f2bf(S);
    if (c + 1 < CC)
      S = pl[(size_t)(b * CC + c + 1) * DD + i] * S +
          sf[((size_t)(b * CC + c) << 14) + ij];
  }
}

// =====================================================================
// K5: per-(b,chunk): A = (Q⊙P)(K/P)ᵀ causal-masked; O = A·V + Q·Sp.
// Only LDS use: 8KB swizzled exchange of A (C-frag -> A-frag).
// =====================================================================
__global__ __launch_bounds__(128) void k_chunk_out(
    const __bf16* __restrict__ qb, const __bf16* __restrict__ qtb,
    const __bf16* __restrict__ ktb, const __bf16* __restrict__ vb,
    const __bf16* __restrict__ spb, __bf16* __restrict__ ob) {
  __shared__ __bf16 As[4096];                       // [mt(4)][kt(2)][lane][16]
  const int lane = threadIdx.x & 31, wv = threadIdx.x >> 5;   // wv = row tile
  const size_t cb = (size_t)blockIdx.x * 8192;
  const size_t sb = (size_t)blockIdx.x << 14;

  v16bf aQt[4], aQ[4];
#pragma unroll
  for (int kt = 0; kt < 4; ++kt) {
    aQt[kt] = frag_ld(qtb + cb + ((wv * 4 + kt) * 32 + lane) * 16);
    aQ[kt]  = frag_ld(qb  + cb + ((wv * 4 + kt) * 32 + lane) * 16);
  }
  // phase A: masked A = Q̃ K̃ᵀ, written to LDS already in A-frag layout
#pragma unroll
  for (int st = 0; st < 4; ++st) {
    v8f acc = {};
#pragma unroll
    for (int kt = 0; kt < 4; ++kt)
      acc = wmma_bf16(aQt[kt], frag_ld(ktb + cb + ((kt * 4 + st) * 32 + lane) * 16), acc);
#pragma unroll
    for (int r = 0; r < 8; ++r) {
      int m = wv * 16 + r + ((lane & 16) ? 8 : 0);  // row t in [0,64)
      int s = st * 16 + (lane & 15);                // col s in [0,64)
      int hi, idx; kdec(s & 31, hi, idx);
      As[((wv * 2 + (s >> 5)) * 32 + ((hi << 4) | (m & 15))) * 16 + idx] =
          f2bf((s <= m) ? acc[r] : 0.f);
    }
  }
  __syncthreads();
  v16bf aA[2];
#pragma unroll
  for (int kt = 0; kt < 2; ++kt)
    aA[kt] = frag_ld(&As[((wv * 2 + kt) * 32 + lane) * 16]);
  // phase B: O = Q·Sp + A·V
#pragma unroll
  for (int nt = 0; nt < 8; ++nt) {
    v8f acc = {};
#pragma unroll
    for (int kt = 0; kt < 4; ++kt)
      acc = wmma_bf16(aQ[kt], frag_ld(spb + sb + ((kt * 8 + nt) * 32 + lane) * 16), acc);
#pragma unroll
    for (int kt = 0; kt < 2; ++kt)
      acc = wmma_bf16(aA[kt], frag_ld(vb + cb + ((kt * 8 + nt) * 32 + lane) * 16), acc);
#pragma unroll
    for (int r = 0; r < 8; ++r) {
      int row = wv * 16 + r + ((lane & 16) ? 8 : 0);
      ob[(size_t)blockIdx.x * LL * DD + (size_t)row * DD + nt * 16 + (lane & 15)] =
          f2bf(acc[r]);
    }
  }
}

// =====================================================================
// K6: final GEMM out = O @ Wo^T (f32 out); A rows straight from bf16 global.
// =====================================================================
__global__ __launch_bounds__(256) void k_out(
    const __bf16* __restrict__ ob, const __bf16* __restrict__ wosw,
    float* __restrict__ out) {
  const int lane = threadIdx.x & 31, wv = threadIdx.x >> 5;
  const size_t m0 = (size_t)blockIdx.x * 128;
  const __bf16* row = ob + (m0 + wv * 16 + (lane & 15)) * DD;
  v16bf a4[4];
#pragma unroll
  for (int kt = 0; kt < 4; ++kt) a4[kt] = fragA_bf16_row(row, kt * 32, lane);
#pragma unroll
  for (int nt = 0; nt < 8; ++nt) {
    v8f acc = {};
#pragma unroll
    for (int kt = 0; kt < 4; ++kt)
      acc = wmma_bf16(a4[kt], frag_ld(wosw + ((kt * 8 + nt) * 32 + lane) * 16), acc);
#pragma unroll
    for (int r = 0; r < 8; ++r) {
      int rrow = wv * 16 + r + ((lane & 16) ? 8 : 0);
      out[(m0 + rrow) * DD + nt * 16 + (lane & 15)] = acc[r];
    }
  }
}

// =====================================================================
extern "C" void kernel_launch(void* const* d_in, const int* in_sizes, int n_in,
                              void* d_out, int out_size, void* d_ws, size_t ws_size,
                              hipStream_t stream) {
  const float* x  = (const float*)d_in[0];
  const float* Wq = (const float*)d_in[1];
  const float* Wk = (const float*)d_in[2];
  const float* Wv = (const float*)d_in[3];
  const float* Wo = (const float*)d_in[4];
  const float* Wd = (const float*)d_in[5];
  float* outp = (float*)d_out;

  char* ws = (char*)d_ws;
  size_t off = 0;
  auto alloc = [&](size_t bytes) -> void* {
    void* p = ws + off;
    off = (off + bytes + 255) & ~(size_t)255;
    return p;
  };
  const size_t SZ_F32  = (size_t)MT * DD * sizeof(float);        // 16 MB
  const size_t SZ_BF16 = (size_t)MT * DD * sizeof(__bf16);       //  8 MB
  const size_t SZ_PL   = (size_t)NBC * DD * sizeof(float);
  const size_t SZ_S    = (size_t)NBC * DD * DD * sizeof(float);  // 32 MB
  const size_t SZ_SB   = (size_t)NBC * DD * DD * sizeof(__bf16); // 16 MB

  __bf16* wsw = (__bf16*)alloc(5 * 16384 * sizeof(__bf16));
  float*  qf  = (float*)alloc(SZ_F32);
  float*  kf  = (float*)alloc(SZ_F32);
  float*  vf  = (float*)alloc(SZ_F32);
  float*  wf  = (float*)alloc(SZ_F32);
  __bf16* qb  = (__bf16*)alloc(SZ_BF16);
  __bf16* qtb = (__bf16*)alloc(SZ_BF16);
  __bf16* ktb = (__bf16*)alloc(SZ_BF16);
  __bf16* khb = (__bf16*)alloc(SZ_BF16);
  __bf16* vb  = (__bf16*)alloc(SZ_BF16);
  float*  pl  = (float*)alloc(SZ_PL);
  float*  sf  = (float*)alloc(SZ_S);
  __bf16* spb = (__bf16*)alloc(SZ_SB);
  __bf16* ob  = (__bf16*)alloc(SZ_BF16);

  k_swizw<<<dim3(5), 256, 0, stream>>>(Wq, Wk, Wv, Wd, Wo, wsw);
  k_proj<<<dim3(MT / 128, 4), 256, 0, stream>>>(x, wsw, qf, kf, vf, wf);
  k_decay<<<dim3(NBC), 128, 0, stream>>>(qf, kf, vf, wf, qb, qtb, ktb, khb, vb, pl);
  k_chunk_state<<<dim3(NBC), 256, 0, stream>>>(khb, vb, sf);
  k_inter<<<dim3((BB * DD * DD) / 256), 256, 0, stream>>>(sf, pl, spb);
  k_chunk_out<<<dim3(NBC), 128, 0, stream>>>(qb, qtb, ktb, vb, spb, ob);
  k_out<<<dim3(MT / 128), 256, 0, stream>>>(ob, wsw + 4 * 16384, outp);
}
